// mha_22660247453920
// MI455X (gfx1250) — compile-verified, matrix operand order checked by reading, not run on hardware
//
#include <hip/hip_runtime.h>
#include <hip/hip_bf16.h>

// ---------------------------------------------------------------------------
// MHA forward for gfx1250 (MI455X): bf16 WMMA + TDM (tensor_load_to_lds)
// staged GEMMs + flash-style attention.
// ---------------------------------------------------------------------------

typedef __attribute__((ext_vector_type(16))) __bf16   v16bf;
typedef __attribute__((ext_vector_type(8)))  __bf16   v8bf;
typedef __attribute__((ext_vector_type(8)))  float    v8f;
typedef __attribute__((ext_vector_type(4)))  unsigned v4u;
typedef __attribute__((ext_vector_type(8)))  int      v8i;
typedef __attribute__((ext_vector_type(4)))  int      v4i;

#define B_SZ    2
#define S_LEN   2048
#define D_MODEL 1024
#define NHEADS  16
#define HDIM    64
#define M_ROWS  (B_SZ * S_LEN)   // 4096

// float -> bf16, round-to-nearest-even
__device__ __forceinline__ __bf16 f2bf(float f) {
  unsigned u = __builtin_bit_cast(unsigned, f);
  unsigned r = (u + 0x7FFFu + ((u >> 16) & 1u)) >> 16;
  unsigned short h = (unsigned short)r;
  return __builtin_bit_cast(__bf16, h);
}

// Load one 16x32 bf16 WMMA fragment (A or B layout per CDNA5 ISA):
// lane(half, idx): elems rp[half*8 + i] and rp[16 + half*8 + i], i=0..7.
__device__ __forceinline__ v16bf frag_ld(const __bf16* rp, int half) {
  v8bf lo = *reinterpret_cast<const v8bf*>(rp + half * 8);
  v8bf hi = *reinterpret_cast<const v8bf*>(rp + 16 + half * 8);
  v16bf f;
#pragma unroll
  for (int i = 0; i < 8; ++i) { f[i] = lo[i]; f[8 + i] = hi[i]; }
  return f;
}

__device__ __forceinline__ v8f wmma_bf16(v16bf a, v16bf b, v8f c) {
  return __builtin_amdgcn_wmma_f32_16x16x32_bf16(
      /*neg_a=*/false, a, /*neg_b=*/false, b,
      /*c_mod=*/(short)0, c, /*reuse_a=*/false, /*reuse_b=*/false);
}

// ---------------------------------------------------------------------------
// TDM: DMA a (rows x 32) bf16 tile (row stride = kstride elements) from a
// row-major global matrix into LDS (packed rows x 32), via Tensor Data Mover.
// D# built per CDNA5 ISA (2D tensor, groups 2/3 zero).
// Issued once per wave (EXEC ignored by TDM); tracked with TENSORcnt.
// 6-arg builtin form (clang-23 / therock-10.0 headers).
// ---------------------------------------------------------------------------
__device__ __forceinline__ void tdm_load_tile(const __bf16* gtile,
                                              unsigned lds_byte_off,
                                              int rows, unsigned kstride) {
  unsigned long long ga = (unsigned long long)(uintptr_t)gtile;
  v4u g0;
  g0[0] = 1u;                                             // count=1, user mode
  g0[1] = lds_byte_off;                                   // lds_addr
  g0[2] = (unsigned)ga;                                   // global_addr[31:0]
  g0[3] = (unsigned)((ga >> 32) & 0x01FFFFFFu) | (2u << 30);  // [56:32]|type=2
  const unsigned td0 = kstride;                           // tensor_dim0
  const unsigned td1 = 1u << 20;                          // tensor_dim1 (no OOB)
  v8i g1;
  g1[0] = (int)(1u << 16);                                // data_size=1 (2B)
  g1[1] = (int)((td0 & 0xFFFFu) << 16);                   // td0[15:0] @63:48
  g1[2] = (int)((td0 >> 16) | ((td1 & 0xFFFFu) << 16));   // td0 hi | td1 lo
  g1[3] = (int)((td1 >> 16) | (32u << 16));               // td1 hi | tile_dim0=32
  g1[4] = (int)(rows & 0xFFFF);                           // tile_dim1=rows
  g1[5] = (int)kstride;                                   // dim0_stride[31:0]
  g1[6] = 0;                                              // dim0_stride hi, dim1_stride lo
  g1[7] = 0;
  v4i z4 = {0, 0, 0, 0};
  v8i z8 = {0, 0, 0, 0, 0, 0, 0, 0};
  __builtin_amdgcn_tensor_load_to_lds(g0, g1, z4, z4, z8, 0);
}

// ---------------------------------------------------------------------------
// fp32 -> bf16 conversion
// ---------------------------------------------------------------------------
__global__ void cvt_f32_bf16(const float* __restrict__ in,
                             __bf16* __restrict__ out, int n) {
  for (int i = blockIdx.x * blockDim.x + threadIdx.x; i < n;
       i += gridDim.x * blockDim.x)
    out[i] = f2bf(in[i]);
}

// ---------------------------------------------------------------------------
// C[M,N] = A[M,K] * W[N,K]^T   (bf16 in, f32 WMMA accumulate)
// TDM double-buffered LDS staging: block tile 64(M) x 128(N), K step 32.
// 8 waves = 2(M) x 4(N); wave tile 32x32; A tile shared by 4 waves, W by 2.
// MODE 0: out bf16  [(b*NH+h)*S + s][HD] + d   (Q/K head layout)
// MODE 1: out bf16  [(b*NH+h)*HD + d][S] + s   (V transposed layout)
// MODE 2: out f32   [m*N + n] = acc + bias[n]  (final projection)
// Grid: (N/128, M/64).
// ---------------------------------------------------------------------------
template <int MODE>
__global__ __launch_bounds__(256) void gemm_nt(
    const __bf16* __restrict__ A, const __bf16* __restrict__ W,
    const float* __restrict__ bias, void* __restrict__ outp,
    int M, int N, int K) {
  __shared__ alignas(16) __bf16 ldsA[2][64][32];    // 2 x 4 KB
  __shared__ alignas(16) __bf16 ldsW[2][128][32];   // 2 x 8 KB

  const int lane = threadIdx.x & 31;
  const int wave = threadIdx.x >> 5;
  const int half = lane >> 4;
  const int idx  = lane & 15;
  const int wm = wave & 1;
  const int wn = wave >> 1;
  const int mBase = blockIdx.y * 64;
  const int nBase = blockIdx.x * 128;

  v8f acc[2][2] = {};
  const int nk = K / 32;

  // Prologue: wave 0 launches TDM for the first two K-steps.
  if (wave == 0) {
    tdm_load_tile(A + (size_t)mBase * K, (unsigned)(uintptr_t)&ldsA[0][0][0],
                  64, (unsigned)K);
    tdm_load_tile(W + (size_t)nBase * K, (unsigned)(uintptr_t)&ldsW[0][0][0],
                  128, (unsigned)K);
    if (nk > 1) {
      tdm_load_tile(A + (size_t)mBase * K + 32,
                    (unsigned)(uintptr_t)&ldsA[1][0][0], 64, (unsigned)K);
      tdm_load_tile(W + (size_t)nBase * K + 32,
                    (unsigned)(uintptr_t)&ldsW[1][0][0], 128, (unsigned)K);
    }
  }

  for (int s = 0; s < nk; ++s) {
    const int cur = s & 1;
    if (wave == 0) {
      // TDM done-signals return in order: <=2 outstanding => step s resident.
      if (s + 1 < nk) __builtin_amdgcn_s_wait_tensorcnt(2);
      else            __builtin_amdgcn_s_wait_tensorcnt(0);
    }
    __syncthreads();   // publish buffer `cur` to all waves

    v16bf a0 = frag_ld(&ldsA[cur][wm * 32 + idx][0], half);
    v16bf a1 = frag_ld(&ldsA[cur][wm * 32 + 16 + idx][0], half);
    v16bf b0 = frag_ld(&ldsW[cur][wn * 32 + idx][0], half);
    v16bf b1 = frag_ld(&ldsW[cur][wn * 32 + 16 + idx][0], half);
    acc[0][0] = wmma_bf16(a0, b0, acc[0][0]);
    acc[0][1] = wmma_bf16(a0, b1, acc[0][1]);
    acc[1][0] = wmma_bf16(a1, b0, acc[1][0]);
    acc[1][1] = wmma_bf16(a1, b1, acc[1][1]);

    __syncthreads();   // all waves done reading `cur` before TDM overwrites it
    if (wave == 0 && s + 2 < nk) {
      tdm_load_tile(A + (size_t)mBase * K + (s + 2) * 32,
                    (unsigned)(uintptr_t)&ldsA[cur][0][0], 64, (unsigned)K);
      tdm_load_tile(W + (size_t)nBase * K + (s + 2) * 32,
                    (unsigned)(uintptr_t)&ldsW[cur][0][0], 128, (unsigned)K);
    }
  }

#pragma unroll
  for (int ti = 0; ti < 2; ++ti) {
#pragma unroll
    for (int tj = 0; tj < 2; ++tj) {
#pragma unroll
      for (int r = 0; r < 8; ++r) {
        const int m = mBase + wm * 32 + ti * 16 + half * 8 + r;
        const int n = nBase + wn * 32 + tj * 16 + idx;
        const float v = acc[ti][tj][r];
        if (MODE == 2) {
          ((float*)outp)[(size_t)m * N + n] = v + bias[n];
        } else {
          const int b = m / S_LEN, s2 = m % S_LEN;
          const int h = n / HDIM,  d  = n % HDIM;
          size_t dst;
          if (MODE == 0)
            dst = ((size_t)(b * NHEADS + h) * S_LEN + s2) * HDIM + d;
          else
            dst = ((size_t)(b * NHEADS + h) * HDIM + d) * S_LEN + s2;
          ((__bf16*)outp)[dst] = f2bf(v);
        }
      }
    }
  }
}

// ---------------------------------------------------------------------------
// Causal flash attention. Q,K: [B*NH][S][HD] bf16; Vt: [B*NH][HD][S] bf16.
// One wave per 16-query tile; 8 waves per block; online softmax; 32 keys/step.
// Output O written bf16 as [B*S][D] (heads re-interleaved).
// Grid: (S/16/8 = 16, B*NH = 32), block 256.
// ---------------------------------------------------------------------------
__global__ __launch_bounds__(256) void attn_kernel(
    const __bf16* __restrict__ Q, const __bf16* __restrict__ Kc,
    const __bf16* __restrict__ Vt, __bf16* __restrict__ O) {
  __shared__ alignas(16) __bf16 pst[8][16][32];  // per-wave P staging

  const int lane = threadIdx.x & 31;
  const int wave = threadIdx.x >> 5;
  const int half = lane >> 4;
  const int idx  = lane & 15;
  const int bh    = blockIdx.y;                // b*NH + h
  const int qtile = blockIdx.x * 8 + wave;     // 0..127
  const int qb    = qtile * 16;

  const __bf16* Qh = Q  + (size_t)bh * S_LEN * HDIM;
  const __bf16* Kh = Kc + (size_t)bh * S_LEN * HDIM;
  const __bf16* Vh = Vt + (size_t)bh * HDIM * S_LEN;

  v16bf aq0 = frag_ld(Qh + (size_t)(qb + idx) * HDIM, half);
  v16bf aq1 = frag_ld(Qh + (size_t)(qb + idx) * HDIM + 32, half);

  v8f o[4] = {};
  float mrow[8], lrow[8];
#pragma unroll
  for (int r = 0; r < 8; ++r) { mrow[r] = -1e30f; lrow[r] = 0.f; }

  const int nsteps = (qtile + 2) >> 1;  // ceil((qtile+1)/2) key tiles of 32
  for (int st = 0; st < nsteps; ++st) {
    const int tb = st * 32;

    v8f s[2];
#pragma unroll
    for (int tt = 0; tt < 2; ++tt) {
      const int kb = tb + tt * 16;
      v16bf bk0 = frag_ld(Kh + (size_t)(kb + idx) * HDIM, half);
      v16bf bk1 = frag_ld(Kh + (size_t)(kb + idx) * HDIM + 32, half);
      v8f z = {};
      z     = wmma_bf16(aq1, bk1, z);
      s[tt] = wmma_bf16(aq0, bk0, z);
    }

    float lm[8];
#pragma unroll
    for (int r = 0; r < 8; ++r) {
      const int q = qb + half * 8 + r;
#pragma unroll
      for (int tt = 0; tt < 2; ++tt) {
        const int t = tb + tt * 16 + idx;
        float v = s[tt][r] * 0.125f;  // 1/sqrt(64)
        if (t > q) v = -1e30f;
        s[tt][r] = v;
      }
      lm[r] = fmaxf(s[0][r], s[1][r]);
    }
#pragma unroll
    for (int mk = 1; mk < 16; mk <<= 1)
#pragma unroll
      for (int r = 0; r < 8; ++r)
        lm[r] = fmaxf(lm[r], __shfl_xor(lm[r], mk, 32));

    float rs[8];
#pragma unroll
    for (int r = 0; r < 8; ++r) {
      const float mn = fmaxf(mrow[r], lm[r]);
      const float sf = __expf(mrow[r] - mn);
      mrow[r] = mn;
      const float p0 = __expf(s[0][r] - mn);
      const float p1 = __expf(s[1][r] - mn);
      pst[wave][half * 8 + r][idx]      = f2bf(p0);
      pst[wave][half * 8 + r][16 + idx] = f2bf(p1);
      rs[r] = p0 + p1;
      o[0][r] *= sf; o[1][r] *= sf; o[2][r] *= sf; o[3][r] *= sf;
      lrow[r] *= sf;
    }
#pragma unroll
    for (int mk = 1; mk < 16; mk <<= 1)
#pragma unroll
      for (int r = 0; r < 8; ++r)
        rs[r] += __shfl_xor(rs[r], mk, 32);
#pragma unroll
    for (int r = 0; r < 8; ++r) lrow[r] += rs[r];

    // wave-local LDS relayout (C-layout -> A-fragment layout)
    asm volatile("s_wait_dscnt 0" ::: "memory");
    v16bf ap = frag_ld(&pst[wave][idx][0], half);

#pragma unroll
    for (int j = 0; j < 4; ++j) {
      v16bf bv = frag_ld(Vh + (size_t)(j * 16 + idx) * S_LEN + tb, half);
      o[j] = wmma_bf16(ap, bv, o[j]);
    }
  }

  const int b = bh >> 4;
  const int h = bh & 15;
#pragma unroll
  for (int j = 0; j < 4; ++j)
#pragma unroll
    for (int r = 0; r < 8; ++r) {
      const int q = qb + half * 8 + r;
      const int d = j * 16 + idx;
      O[(size_t)(b * S_LEN + q) * D_MODEL + h * HDIM + d] =
          f2bf(o[j][r] / lrow[r]);
    }
}

// ---------------------------------------------------------------------------
// Launch
// ---------------------------------------------------------------------------
extern "C" void kernel_launch(void* const* d_in, const int* in_sizes, int n_in,
                              void* d_out, int out_size, void* d_ws,
                              size_t ws_size, hipStream_t stream) {
  const float* x    = (const float*)d_in[0];
  const float* wq   = (const float*)d_in[1];
  const float* wk   = (const float*)d_in[2];
  const float* wv   = (const float*)d_in[3];
  const float* wo_w = (const float*)d_in[4];
  const float* wo_b = (const float*)d_in[5];

  char* ws = (char*)d_ws;
  const size_t MB = 1u << 20;
  __bf16* xb  = (__bf16*)(ws + 0);        // 8 MB  [4096][1024]
  __bf16* wqb = (__bf16*)(ws + 8 * MB);   // 2 MB
  __bf16* wkb = (__bf16*)(ws + 10 * MB);  // 2 MB
  __bf16* wvb = (__bf16*)(ws + 12 * MB);  // 2 MB
  __bf16* wob = (__bf16*)(ws + 14 * MB);  // 2 MB
  __bf16* Qb  = (__bf16*)(ws + 16 * MB);  // 8 MB  [B*NH][S][HD]
  __bf16* Kb  = (__bf16*)(ws + 24 * MB);  // 8 MB  [B*NH][S][HD]
  __bf16* VTb = (__bf16*)(ws + 32 * MB);  // 8 MB  [B*NH][HD][S]
  __bf16* Ob  = (__bf16*)(ws + 40 * MB);  // 8 MB  [4096][1024]

  const int nX = M_ROWS * D_MODEL;
  const int nW = D_MODEL * D_MODEL;

  cvt_f32_bf16<<<2048, 256, 0, stream>>>(x,    xb,  nX);
  cvt_f32_bf16<<<1024, 256, 0, stream>>>(wq,   wqb, nW);
  cvt_f32_bf16<<<1024, 256, 0, stream>>>(wk,   wkb, nW);
  cvt_f32_bf16<<<1024, 256, 0, stream>>>(wv,   wvb, nW);
  cvt_f32_bf16<<<1024, 256, 0, stream>>>(wo_w, wob, nW);

  dim3 ggrid(D_MODEL / 128, M_ROWS / 64);  // (8, 64)
  gemm_nt<0><<<ggrid, 256, 0, stream>>>(xb, wqb, nullptr, Qb,
                                        M_ROWS, D_MODEL, D_MODEL);
  gemm_nt<0><<<ggrid, 256, 0, stream>>>(xb, wkb, nullptr, Kb,
                                        M_ROWS, D_MODEL, D_MODEL);
  gemm_nt<1><<<ggrid, 256, 0, stream>>>(xb, wvb, nullptr, VTb,
                                        M_ROWS, D_MODEL, D_MODEL);

  attn_kernel<<<dim3(S_LEN / 16 / 8, B_SZ * NHEADS), 256, 0, stream>>>(
      Qb, Kb, VTb, Ob);

  gemm_nt<2><<<ggrid, 256, 0, stream>>>(Ob, wob, wo_b, d_out,
                                        M_ROWS, D_MODEL, D_MODEL);
}
